// ETFaithfulGraphModel_6708738916961
// MI455X (gfx1250) — compile-verified, matrix operand order checked by reading, not run on hardware
//
#include <hip/hip_runtime.h>
#include <cstdint>
#include <cstddef>

// ---------------- model dimensions ----------------
#define G_    64
#define N_    256
#define T_    257        // tokens per graph (CLS + N)
#define TP_   288        // padded tokens: multiple of 32 for WMMA K-loops
#define D_    128
#define H_    4
#define HD_   32
#define PEK_  16
#define KM_   32
#define IN_   64
#define HID_  256
#define NC_   10
#define NSTEP 8
#define NV_   17         // power-iteration subspace (trivial + 16 PE vectors)
#define PITERS 30

#define PWR_USE_ASYNC 1  // gfx1250 async global->LDS staging in eigensolver

typedef __bf16 bf16_t;
typedef __attribute__((ext_vector_type(16))) __bf16 v16bf;
typedef __attribute__((ext_vector_type(8)))  __bf16 v8bf;
typedef __attribute__((ext_vector_type(8)))  float  v8f;

#define DEVFN static __device__ __forceinline__

// ---------------- wave32 helpers ----------------
DEVFN float wred_sum(float v){
#pragma unroll
  for (int o = 16; o > 0; o >>= 1) v += __shfl_xor(v, o, 32);
  return v;
}
DEVFN float wred_max(float v){
#pragma unroll
  for (int o = 16; o > 0; o >>= 1) v = fmaxf(v, __shfl_xor(v, o, 32));
  return v;
}
DEVFN float geluf(float x){ return 0.5f * x * (1.0f + erff(x * 0.7071067811865476f)); }
DEVFN float softplusf(float x){ return (x > 20.f) ? x : log1pf(expf(x)); }

union FragU { v16bf v; v8bf h[2]; };

// 16x32 bf16 fragment loader matching the CDNA5 WMMA A/B register layout:
// lane L (m = L&15, khalf = L>>4): regs[0..7] = row m, K = khalf*8 .. +7;
// regs[8..15] = row m, K = 16 + khalf*8 .. +7.  Requires ld % 8 == 0.
DEVFN v16bf load_frag(const bf16_t* base, int ld){
  int lane = threadIdx.x & 31;
  int m    = lane & 15;
  int kh   = (lane >> 4) << 3;
  const bf16_t* p = base + (size_t)m * ld + kh;
  FragU u;
  u.h[0] = *(const v8bf*)(p);
  u.h[1] = *(const v8bf*)(p + 16);
  return u.v;
}

DEVFN v8f wmma_bf16(v16bf a, v16bf b, v8f c){
  return __builtin_amdgcn_wmma_f32_16x16x32_bf16(false, a, false, b, (short)0, c, false, false);
}

// ---------------- generic batched WMMA GEMM ----------------
// Each wave computes a 16x32 output tile: one A fragment feeds two accumulators,
// and fragment loads are software-pipelined one K-step ahead so global_load_b128
// of step k+1 overlaps the two v_wmma of step k (partial loadcnt waits).
// C[b] = op( scale * A[b] (MxK) * B[b] (KxN) + bias ), Bt holds B^T (N rows of K).
#define GF_GELU 1
#define GF_ACC  2
#define GF_WF32 4
#define GF_WBF  8

__global__ __launch_bounds__(128) void k_gemm(
    const bf16_t* __restrict__ A, const bf16_t* __restrict__ Bt,
    float* __restrict__ Cf, bf16_t* __restrict__ Cbf,
    const float* __restrict__ bias, const float* __restrict__ scalep,
    int K, int lda, int ldb, int ldc,
    int batch2, long sA, long sB, long sC, long sA2, long sB2, long sC2,
    int ntm, int ntn2, int flags, int total)
{
  int wid = blockIdx.x * 4 + (threadIdx.x >> 5);
  if (wid >= total) return;
  int tpm = ntm * ntn2;
  int b  = wid / tpm, t = wid % tpm;
  int b1 = b / batch2, b2 = b % batch2;
  int ti = t / ntn2, tj = t % ntn2;
  const bf16_t* Ab  = A  + (size_t)b1 * sA + (size_t)b2 * sA2 + (size_t)ti * 16 * lda;
  const bf16_t* Bb0 = Bt + (size_t)b1 * sB + (size_t)b2 * sB2 + (size_t)tj * 32 * ldb;
  const bf16_t* Bb1 = Bb0 + (size_t)16 * ldb;
  size_t coff = (size_t)b1 * sC + (size_t)b2 * sC2 + (size_t)ti * 16 * ldc + (size_t)tj * 32;
  int lane = threadIdx.x & 31;
  int cn = lane & 15, cm0 = (lane >> 4) * 8;
  v8f acc0 = {}, acc1 = {};
  if (flags & GF_ACC){
#pragma unroll
    for (int v = 0; v < 8; ++v){
      acc0[v] = Cf[coff + (size_t)(cm0 + v) * ldc + cn];
      acc1[v] = Cf[coff + (size_t)(cm0 + v) * ldc + cn + 16];
    }
  }
  // pipelined fragment loads: fetch k+1 while multiplying k
  v16bf fa  = load_frag(Ab,  lda);
  v16bf fb0 = load_frag(Bb0, ldb);
  v16bf fb1 = load_frag(Bb1, ldb);
  for (int kk = 32; kk < K; kk += 32){
    __builtin_prefetch(Ab + (size_t)kk + 32, 0, 1);
    v16bf na  = load_frag(Ab  + kk, lda);
    v16bf nb0 = load_frag(Bb0 + kk, ldb);
    v16bf nb1 = load_frag(Bb1 + kk, ldb);
    acc0 = wmma_bf16(fa, fb0, acc0);
    acc1 = wmma_bf16(fa, fb1, acc1);
    fa = na; fb0 = nb0; fb1 = nb1;
  }
  acc0 = wmma_bf16(fa, fb0, acc0);
  acc1 = wmma_bf16(fa, fb1, acc1);

  float sc = scalep ? *scalep : 1.f;
#pragma unroll
  for (int v = 0; v < 8; ++v){
    float x0 = acc0[v] * sc;
    float x1 = acc1[v] * sc;
    if (bias){
      x0 += bias[tj * 32 + cn];
      x1 += bias[tj * 32 + cn + 16];
    }
    if (flags & GF_GELU){ x0 = geluf(x0); x1 = geluf(x1); }
    size_t o = coff + (size_t)(cm0 + v) * ldc + cn;
    if (flags & GF_WF32){ Cf[o] = x0;  Cf[o + 16] = x1; }
    if (flags & GF_WBF) { Cbf[o] = (bf16_t)x0; Cbf[o + 16] = (bf16_t)x1; }
  }
}

// ---------------- small conversion / prep kernels ----------------
__global__ void k_cvt(const float* __restrict__ s, bf16_t* __restrict__ d, size_t n){
  for (size_t i = (size_t)blockIdx.x * blockDim.x + threadIdx.x; i < n;
       i += (size_t)gridDim.x * blockDim.x) d[i] = (bf16_t)s[i];
}
// s[rows][cols] -> d[cols][rows]
__global__ void k_cvt_t(const float* __restrict__ s, bf16_t* __restrict__ d, int rows, int cols){
  size_t n = (size_t)rows * cols;
  for (size_t i = (size_t)blockIdx.x * blockDim.x + threadIdx.x; i < n;
       i += (size_t)gridDim.x * blockDim.x){
    int rr = (int)(i % rows); int cc = (int)(i / rows);
    d[i] = (bf16_t)s[(size_t)rr * cols + cc];
  }
}
__global__ __launch_bounds__(128) void k_km(const float* __restrict__ Bm,
                                            const float* __restrict__ Wkm,
                                            bf16_t* __restrict__ Kmb){
  int k = blockIdx.x, d = threadIdx.x;
  float acc = 0.f;
  for (int e = 0; e < D_; ++e) acc += Bm[(size_t)k * D_ + e] * Wkm[(size_t)e * D_ + d];
  Kmb[(size_t)k * D_ + d] = (bf16_t)acc;
}
__global__ void k_scalars(const float* la, const float* lm, const float* ba,
                          const float* bm, const float* el, float* s){
  if (threadIdx.x == 0 && blockIdx.x == 0){
    float l0 = softplusf(*la), b0 = fminf(softplusf(*ba), 5.f);
    float l1 = softplusf(*lm), b1 = fminf(softplusf(*bm), 5.f);
    s[0] = l0; s[1] = b0; s[2] = l1; s[3] = b1;
    s[4] = 0.25f / (1.f + expf(-(*el)));      // eta
    s[5] = l0 / sqrtf(32.f);                  // lam_att / sqrt(HD)
    s[6] = l1 / sqrtf(128.f);                 // lam_m / sqrt(D)
  }
}

// ---------------- graph structure ----------------
DEVFN bool edge_fn(const unsigned char* adj, int g, int i, int j){
  if (i == j) return false;
  if (i == 0 || j == 0) return true;                 // CLS connects everything
  return adj[((size_t)g * N_ + (i - 1)) * N_ + (j - 1)] != 0;
}
__global__ __launch_bounds__(256) void k_degree(const unsigned char* __restrict__ adj,
                                                float* __restrict__ deg){
  int w = blockIdx.x * 8 + (threadIdx.x >> 5);
  if (w >= G_ * T_) return;
  int g = w / T_, t = w % T_;
  int lane = threadIdx.x & 31;
  float dv;
  if (t == 0) dv = (float)N_;
  else {
    float s = 0.f;
    const unsigned char* row = adj + ((size_t)g * N_ + (t - 1)) * N_;
    for (int j = lane; j < N_; j += 32) s += row[j] ? 1.f : 0.f;
    s = wred_sum(s);
    dv = 1.f + s;                                    // + CLS edge
  }
  if (lane == 0) deg[(size_t)g * T_ + t] = dv;
}
// M = I + D^-1/2 A D^-1/2 = 2I - L ; padded entries zero.
__global__ void k_build_M(const unsigned char* __restrict__ adj,
                          const float* __restrict__ deg, bf16_t* __restrict__ Mm){
  size_t n = (size_t)G_ * TP_ * TP_;
  for (size_t idx = (size_t)blockIdx.x * blockDim.x + threadIdx.x; idx < n;
       idx += (size_t)gridDim.x * blockDim.x){
    int j = (int)(idx % TP_); size_t r = idx / TP_;
    int i = (int)(r % TP_);   int g = (int)(r / TP_);
    float m = 0.f;
    if (i < T_ && j < T_){
      if (i == j) m = 1.f;
      else if (edge_fn(adj, g, i, j))
        m = rsqrtf(deg[(size_t)g * T_ + i] * deg[(size_t)g * T_ + j]);
    }
    Mm[idx] = (bf16_t)m;
  }
}

// ---------------- Laplacian-PE eigensolver ----------------
// Orthogonal (block power) iteration on M = 2I - L, entirely in LDS.
// CDNA5's 320 KB WGP LDS holds M (288x288 bf16 = 166 KB) + V,Y (f32) + V^T (bf16).
DEVFN float blk_sum256(float v, float* red){
  v = wred_sum(v);
  int lane = threadIdx.x & 31, wv = threadIdx.x >> 5;
  if (lane == 0) red[wv] = v;
  __syncthreads();
  float s = 0.f;
#pragma unroll
  for (int i = 0; i < 8; ++i) s += red[i];
  __syncthreads();
  return s;
}
#define PWR_SMEM ((size_t)TP_*TP_*2 + (size_t)TP_*32*4*2 + (size_t)32*TP_*2 + 64)

__global__ __launch_bounds__(256) void k_power(const bf16_t* __restrict__ Mg,
                                               float* __restrict__ Vout){
  extern __shared__ char smem[];
  bf16_t* Ml  = (bf16_t*)smem;                               // [TP_][TP_]
  float*  Vf  = (float*)(smem + (size_t)TP_ * TP_ * 2);      // [TP_][32]
  float*  Yf  = Vf + (size_t)TP_ * 32;                       // [TP_][32]
  bf16_t* Vbt = (bf16_t*)(Yf + (size_t)TP_ * 32);            // [32][TP_] (V^T)
  float*  red = (float*)(Vbt + (size_t)32 * TP_);
  int g = blockIdx.x, tid = threadIdx.x;
  int lane = tid & 31, wv = tid >> 5;
  const bf16_t* Msrc = Mg + (size_t)g * TP_ * TP_;

  // stage M into LDS (async DMA path on gfx1250)
#if PWR_USE_ASYNC
  for (int c = tid; c < (TP_ * TP_) / 8; c += 256){
    const bf16_t* gp = Msrc + (size_t)c * 8;
    unsigned lo = (unsigned)(uintptr_t)(Ml + (size_t)c * 8);
    asm volatile("global_load_async_to_lds_b128 %0, %1, off"
                 :: "v"(lo), "v"(gp) : "memory");
  }
  asm volatile("s_wait_asynccnt 0" ::: "memory");
#else
  for (int c = tid; c < (TP_ * TP_) / 8; c += 256)
    *(v8bf*)(Ml + (size_t)c * 8) = *(const v8bf*)(Msrc + (size_t)c * 8);
#endif
  __syncthreads();

  // init subspace: col0 ~ trivial direction, others deterministic pseudo-random
  for (int idx = tid; idx < TP_ * 32; idx += 256){
    int t = idx / 32, c = idx % 32;
    float v = 0.f;
    if (t < T_ && c < NV_)
      v = (c == 0) ? 1.f : sinf(0.37f * (float)(t + 1) * (float)(7 * c + 3));
    Vf[idx] = v;
  }
  __syncthreads();

  for (int it = 0; it <= PITERS; ++it){
    // modified Gram-Schmidt on the NV_ columns
    for (int c = 0; c < NV_; ++c){
      for (int p = 0; p < c; ++p){
        float part = 0.f;
        for (int t = tid; t < TP_; t += 256) part += Vf[t * 32 + c] * Vf[t * 32 + p];
        float dot = blk_sum256(part, red);
        for (int t = tid; t < TP_; t += 256) Vf[t * 32 + c] -= dot * Vf[t * 32 + p];
        __syncthreads();
      }
      float part = 0.f;
      for (int t = tid; t < TP_; t += 256){ float x = Vf[t * 32 + c]; part += x * x; }
      float nn = blk_sum256(part, red);
      float inv = rsqrtf(fmaxf(nn, 1e-12f));
      for (int t = tid; t < TP_; t += 256) Vf[t * 32 + c] *= inv;
      __syncthreads();
    }
    if (it == PITERS) break;
    // V^T in bf16 for the WMMA B operand
    for (int idx = tid; idx < TP_ * 32; idx += 256){
      int t = idx / 32, c = idx % 32;
      Vbt[(size_t)c * TP_ + t] = (bf16_t)Vf[idx];
    }
    __syncthreads();
    // Y = M * V : 18 row tiles, each wave computes 16x32 with one shared A fragment
    for (int ti = wv; ti < 18; ti += 8){
      v8f acc0 = {}, acc1 = {};
      for (int kk = 0; kk < TP_; kk += 32){
        v16bf fa  = load_frag(Ml  + (size_t)ti * 16 * TP_ + kk, TP_);
        v16bf fb0 = load_frag(Vbt + kk, TP_);
        v16bf fb1 = load_frag(Vbt + (size_t)16 * TP_ + kk, TP_);
        acc0 = wmma_bf16(fa, fb0, acc0);
        acc1 = wmma_bf16(fa, fb1, acc1);
      }
      int cn = lane & 15, cm0 = (lane >> 4) * 8;
#pragma unroll
      for (int v = 0; v < 8; ++v){
        Yf[(size_t)(ti * 16 + cm0 + v) * 32 + cn]      = acc0[v];
        Yf[(size_t)(ti * 16 + cm0 + v) * 32 + 16 + cn] = acc1[v];
      }
    }
    __syncthreads();
    for (int idx = tid; idx < TP_ * 32; idx += 256) Vf[idx] = Yf[idx];
    __syncthreads();
  }
  for (int idx = tid; idx < TP_ * 32; idx += 256)
    Vout[(size_t)g * TP_ * 32 + idx] = Vf[idx];
}

// ---------------- encoder LN + PE assembly ----------------
__global__ __launch_bounds__(256) void k_assemble(
    const float* __restrict__ z, const float* __restrict__ eg, const float* __restrict__ eb,
    const float* __restrict__ cls, const float* __restrict__ Vvec,
    const float* __restrict__ peW, const float* __restrict__ peb, float* __restrict__ xa)
{
  int w = blockIdx.x * 8 + (threadIdx.x >> 5);
  if (w >= G_ * T_) return;
  int g = w / T_, t = w % T_;
  int lane = threadIdx.x & 31;
  float base[4];
  if (t == 0){
#pragma unroll
    for (int i = 0; i < 4; ++i) base[i] = cls[lane + 32 * i];
  } else {
    const float* zr = z + ((size_t)g * N_ + (t - 1)) * D_;
    float v[4], s = 0.f;
#pragma unroll
    for (int i = 0; i < 4; ++i){ v[i] = zr[lane + 32 * i]; s += v[i]; }
    s = wred_sum(s); float m = s * (1.f / D_);
    float q = 0.f;
#pragma unroll
    for (int i = 0; i < 4; ++i){ float d = v[i] - m; q += d * d; }
    q = wred_sum(q);
    float rstd = rsqrtf(q * (1.f / D_) + 1e-5f);
#pragma unroll
    for (int i = 0; i < 4; ++i){
      int d = lane + 32 * i;
      base[i] = (v[i] - m) * rstd * eg[d] + eb[d];
    }
  }
  const float* pv = Vvec + ((size_t)g * TP_ + t) * 32;
  float* o = xa + ((size_t)g * TP_ + t) * D_;
#pragma unroll
  for (int i = 0; i < 4; ++i){
    int d = lane + 32 * i;
    float acc = 0.f;
    for (int k = 0; k < PEK_; ++k) acc += pv[1 + k] * peW[(size_t)k * D_ + d];
    o[d] = base[i] + acc + peb[d];
  }
}

// ---------------- descent-step kernels ----------------
__global__ __launch_bounds__(256) void k_ln(const float* __restrict__ xa,
                                            const float* __restrict__ lg,
                                            const float* __restrict__ lb,
                                            bf16_t* __restrict__ gbf){
  int w = blockIdx.x * 8 + (threadIdx.x >> 5);
  if (w >= G_ * T_) return;
  int g = w / T_, t = w % T_;
  int lane = threadIdx.x & 31;
  const float* x = xa + ((size_t)g * TP_ + t) * D_;
  bf16_t* o = gbf + ((size_t)g * TP_ + t) * D_;
  float v[4], s = 0.f;
#pragma unroll
  for (int i = 0; i < 4; ++i){ v[i] = x[lane + 32 * i]; s += v[i]; }
  s = wred_sum(s); float m = s * (1.f / D_);
  float q = 0.f;
#pragma unroll
  for (int i = 0; i < 4; ++i){ float d = v[i] - m; q += d * d; }
  q = wred_sum(q);
  float rstd = rsqrtf(q * (1.f / D_) + 1e-5f);
#pragma unroll
  for (int i = 0; i < 4; ++i){
    int d = lane + 32 * i;
    o[d] = (bf16_t)((v[i] - m) * rstd * lg[d] + lb[d]);
  }
}

__global__ __launch_bounds__(256) void k_softmax(const float* __restrict__ S,
                                                 const unsigned char* __restrict__ adj,
                                                 const float* __restrict__ scal,
                                                 bf16_t* __restrict__ P){
  int w = blockIdx.x * 8 + (threadIdx.x >> 5);
  if (w >= G_ * H_ * T_) return;
  int i = w % T_; int r = w / T_;
  int h = r % H_; int g = r / H_;
  int lane = threadIdx.x & 31;
  const float* Sb = S + (size_t)(g * H_ + h) * TP_ * TP_;
  bf16_t* Pb = P + (size_t)(g * H_ + h) * TP_ * TP_ + (size_t)i * TP_;
  float beta = scal[1];
  const float inv_scale = 0.17677669529663687f;   // 1/sqrt(HD)
  float l[9];
  float mx = -3.0e38f;
#pragma unroll
  for (int it = 0; it < 9; ++it){
    int j = lane + 32 * it;
    float v = -3.0e38f;
    if (j < T_ && edge_fn(adj, g, i, j))
      v = beta * (Sb[(size_t)i * TP_ + j] + Sb[(size_t)j * TP_ + i]) * inv_scale;
    l[it] = v; mx = fmaxf(mx, v);
  }
  mx = wred_max(mx);
  float sum = 0.f;
#pragma unroll
  for (int it = 0; it < 9; ++it){
    float e = (l[it] > -1e37f) ? __expf(l[it] - mx) : 0.f;
    l[it] = e; sum += e;
  }
  sum = wred_sum(sum);
  float inv = 1.f / sum;
#pragma unroll
  for (int it = 0; it < 9; ++it){
    int j = lane + 32 * it;
    Pb[j] = (bf16_t)(l[it] * inv);   // pad cols (j>=T_) get 0 since l==0 there
  }
}

__global__ void k_sym(const bf16_t* __restrict__ P, bf16_t* __restrict__ Wsym){
  size_t n = (size_t)G_ * H_ * TP_ * TP_;
  for (size_t idx = (size_t)blockIdx.x * blockDim.x + threadIdx.x; idx < n;
       idx += (size_t)gridDim.x * blockDim.x){
    int j = (int)(idx % TP_); size_t r = idx / TP_;
    int i = (int)(r % TP_);   size_t gh = r / TP_;
    float v = 0.f;
    if (i < T_ && j < T_){
      const bf16_t* Pb = P + gh * TP_ * TP_;
      v = (float)Pb[(size_t)i * TP_ + j] + (float)Pb[(size_t)j * TP_ + i];
    }
    Wsym[idx] = (bf16_t)v;
  }
}

__global__ void k_qk_transpose(const bf16_t* __restrict__ Qb, const bf16_t* __restrict__ Kb,
                               bf16_t* __restrict__ Qtr, bf16_t* __restrict__ Ktr){
  size_t n = (size_t)G_ * H_ * HD_ * TP_;
  for (size_t idx = (size_t)blockIdx.x * blockDim.x + threadIdx.x; idx < n;
       idx += (size_t)gridDim.x * blockDim.x){
    int t  = (int)(idx % TP_); size_t r = idx / TP_;
    int nn = (int)(r % HD_);   r /= HD_;
    int h  = (int)(r % H_);    int g = (int)(r / H_);
    size_t src = ((size_t)g * TP_ + t) * D_ + h * HD_ + nn;
    Qtr[idx] = Qb[src];
    Ktr[idx] = Kb[src];
  }
}

__global__ __launch_bounds__(256) void k_mem(const bf16_t* __restrict__ qmb,
                                             const bf16_t* __restrict__ Kmb,
                                             const float* __restrict__ scal,
                                             bf16_t* __restrict__ dqmb){
  int w = blockIdx.x * 8 + (threadIdx.x >> 5);
  if (w >= G_ * TP_) return;
  int t = w % TP_;
  int lane = threadIdx.x & 31;
  bf16_t* o = dqmb + (size_t)w * D_;
  if (t >= T_){
#pragma unroll
    for (int i = 0; i < 4; ++i) o[lane + 32 * i] = (bf16_t)0.f;
    return;
  }
  const bf16_t* q  = qmb + (size_t)w * D_;
  const bf16_t* km = Kmb + (size_t)lane * D_;
  float lm = 0.f;
  for (int d = 0; d < D_; ++d) lm += (float)q[d] * (float)km[d];
  lm *= 0.08838834764831845f;                 // 1/sqrt(D)
  float logit = scal[3] * lm;
  float mx = wred_max(logit);
  float e  = __expf(logit - mx);
  float s  = wred_sum(e);
  float p  = e / s;
  float c6 = scal[6];
  float acc[4] = {0.f, 0.f, 0.f, 0.f};
  for (int k = 0; k < 32; ++k){
    float pk = __shfl(p, k, 32);
    const bf16_t* kr = Kmb + (size_t)k * D_;
#pragma unroll
    for (int i = 0; i < 4; ++i) acc[i] += pk * (float)kr[lane + 32 * i];
  }
#pragma unroll
  for (int i = 0; i < 4; ++i) o[lane + 32 * i] = (bf16_t)(acc[i] * c6);
}

// LN vector-Jacobian product + energy-descent update with grad/state clipping.
__global__ __launch_bounds__(256) void k_update(float* __restrict__ xa,
                                                const float* __restrict__ dg,
                                                const float* __restrict__ lg,
                                                const float* __restrict__ scal){
  int w = blockIdx.x * 8 + (threadIdx.x >> 5);
  if (w >= G_ * T_) return;
  int g = w / T_, t = w % T_;
  int lane = threadIdx.x & 31;
  float* xr = xa + ((size_t)g * TP_ + t) * D_;
  const float* dr = dg + ((size_t)g * TP_ + t) * D_;
  float x[4], dv[4], gam[4];
  float s = 0.f;
#pragma unroll
  for (int i = 0; i < 4; ++i){
    int d = lane + 32 * i;
    x[i] = xr[d]; dv[i] = dr[d]; gam[i] = lg[d];
    s += x[i];
  }
  s = wred_sum(s); float m = s * (1.f / D_);
  float q = 0.f;
#pragma unroll
  for (int i = 0; i < 4; ++i){ float d = x[i] - m; q += d * d; }
  q = wred_sum(q);
  float rstd = rsqrtf(q * (1.f / D_) + 1e-5f);
  float xh[4], a[4];
  float s1 = 0.f, s2 = 0.f;
#pragma unroll
  for (int i = 0; i < 4; ++i){
    xh[i] = (x[i] - m) * rstd;
    a[i]  = dv[i] * gam[i];
    s1 += a[i]; s2 += a[i] * xh[i];
  }
  s1 = wred_sum(s1) * (1.f / D_);
  s2 = wred_sum(s2) * (1.f / D_);
  float gr[4], gn2 = 0.f;
#pragma unroll
  for (int i = 0; i < 4; ++i){
    float dx = (a[i] - s1 - xh[i] * s2) * rstd;   // d(e)/dx through LN
    gr[i] = x[i] - dx;                             // grad E = x - d(e)/dx
    gn2 += gr[i] * gr[i];
  }
  gn2 = wred_sum(gn2);
  float gn = sqrtf(gn2);
  float cg = fminf(1.f, 1.f / fmaxf(gn, 1e-6f));
  float eta = scal[4];
  float sn2 = 0.f;
#pragma unroll
  for (int i = 0; i < 4; ++i){
    x[i] -= eta * gr[i] * cg;
    sn2 += x[i] * x[i];
  }
  sn2 = wred_sum(sn2);
  float cs = fminf(1.f, 10.f / fmaxf(sqrtf(sn2), 1e-6f));
#pragma unroll
  for (int i = 0; i < 4; ++i) xr[lane + 32 * i] = x[i] * cs;
}

// ---------------- readout ----------------
DEVFN float blk_sum128(float v, float* red){
  v = wred_sum(v);
  int lane = threadIdx.x & 31, wv = threadIdx.x >> 5;
  if (lane == 0) red[wv] = v;
  __syncthreads();
  float s = red[0] + red[1] + red[2] + red[3];
  __syncthreads();
  return s;
}
__global__ __launch_bounds__(128) void k_readout(
    const float* __restrict__ xa, const float* __restrict__ lg, const float* __restrict__ lb,
    const float* __restrict__ W1, const float* __restrict__ B1,
    const float* __restrict__ G1, const float* __restrict__ BB1,
    const float* __restrict__ W2, const float* __restrict__ B2,
    const float* __restrict__ G2, const float* __restrict__ BB2,
    const float* __restrict__ W3, const float* __restrict__ B3, float* __restrict__ out)
{
  __shared__ float r[128], t1[256], t2[128], red[4];
  int g = blockIdx.x, tid = threadIdx.x;
  const float* x = xa + (size_t)g * TP_ * D_;
  float v = x[tid];
  float m = blk_sum128(v, red) * (1.f / 128.f);
  float q = blk_sum128((v - m) * (v - m), red) * (1.f / 128.f);
  r[tid] = (v - m) * rsqrtf(q + 1e-5f) * lg[tid] + lb[tid];
  __syncthreads();
  float o0 = B1[tid], o1 = B1[tid + 128];
  for (int d = 0; d < 128; ++d){
    float rv = r[d];
    o0 += rv * W1[(size_t)d * 256 + tid];
    o1 += rv * W1[(size_t)d * 256 + tid + 128];
  }
  o0 = geluf(o0); o1 = geluf(o1);
  m = blk_sum128(o0 + o1, red) * (1.f / 256.f);
  q = blk_sum128((o0 - m) * (o0 - m) + (o1 - m) * (o1 - m), red) * (1.f / 256.f);
  float rstd = rsqrtf(q + 1e-5f);
  t1[tid]       = (o0 - m) * rstd * G1[tid] + BB1[tid];
  t1[tid + 128] = (o1 - m) * rstd * G1[tid + 128] + BB1[tid + 128];
  __syncthreads();
  float o = B2[tid];
  for (int d = 0; d < 256; ++d) o += t1[d] * W2[(size_t)d * 128 + tid];
  o = geluf(o);
  m = blk_sum128(o, red) * (1.f / 128.f);
  q = blk_sum128((o - m) * (o - m), red) * (1.f / 128.f);
  t2[tid] = (o - m) * rsqrtf(q + 1e-5f) * G2[tid] + BB2[tid];
  __syncthreads();
  if (tid < NC_){
    float acc = B3[tid];
    for (int d = 0; d < 128; ++d) acc += t2[d] * W3[(size_t)d * NC_ + tid];
    out[(size_t)g * NC_ + tid] = acc;
  }
}

// ---------------- host side ----------------
static void gemm_launch(hipStream_t st, const bf16_t* A, const bf16_t* Bt,
                        float* Cf, bf16_t* Cbf, const float* bias, const float* scale,
                        int M, int N, int K, int lda, int ldb, int ldc,
                        int batch, int batch2,
                        long sA, long sB, long sC, long sA2, long sB2, long sC2, int flags)
{
  int ntm = M / 16, ntn2 = N / 32;
  long total = (long)batch * batch2 * ntm * ntn2;
  int blocks = (int)((total + 3) / 4);
  k_gemm<<<blocks, 128, 0, st>>>(A, Bt, Cf, Cbf, bias, scale, K, lda, ldb, ldc,
                                 batch2, sA, sB, sC, sA2, sB2, sC2,
                                 ntm, ntn2, flags, (int)total);
}

extern "C" void kernel_launch(void* const* d_in, const int* in_sizes, int n_in,
                              void* d_out, int out_size, void* d_ws, size_t ws_size,
                              hipStream_t stream)
{
  (void)in_sizes; (void)n_in; (void)out_size;
  const float*         x        = (const float*)d_in[0];
  const unsigned char* adj      = (const unsigned char*)d_in[1];
  const float* enc_W1 = (const float*)d_in[2];
  const float* enc_b1 = (const float*)d_in[3];
  const float* enc_W2 = (const float*)d_in[4];
  const float* enc_b2 = (const float*)d_in[5];
  const float* enc_lg = (const float*)d_in[6];
  const float* enc_lb = (const float*)d_in[7];
  const float* pe_W   = (const float*)d_in[8];
  const float* pe_b   = (const float*)d_in[9];
  const float* cls    = (const float*)d_in[10];
  const float* ln_g   = (const float*)d_in[11];
  const float* ln_b   = (const float*)d_in[12];
  const float* W_Q    = (const float*)d_in[13];
  const float* W_K    = (const float*)d_in[14];
  const float* W_Qm   = (const float*)d_in[15];
  const float* W_Km   = (const float*)d_in[16];
  const float* B_mem  = (const float*)d_in[17];
  const float* lam_a  = (const float*)d_in[18];
  const float* lam_m  = (const float*)d_in[19];
  const float* bet_a  = (const float*)d_in[20];
  const float* bet_m  = (const float*)d_in[21];
  const float* eta_l  = (const float*)d_in[22];
  const float* rd_W1  = (const float*)d_in[23];
  const float* rd_b1  = (const float*)d_in[24];
  const float* rd_g1  = (const float*)d_in[25];
  const float* rd_bb1 = (const float*)d_in[26];
  const float* rd_W2  = (const float*)d_in[27];
  const float* rd_b2  = (const float*)d_in[28];
  const float* rd_g2  = (const float*)d_in[29];
  const float* rd_bb2 = (const float*)d_in[30];
  const float* rd_W3  = (const float*)d_in[31];
  const float* rd_b3  = (const float*)d_in[32];
  float* out = (float*)d_out;

  const size_t GN = (size_t)G_ * N_;
  const size_t GT = (size_t)G_ * TP_;

  char* wsb = (char*)d_ws; size_t off = 0;
  auto take = [&](size_t b) -> char* {
    char* p = wsb + off; off = (off + b + 255) & ~(size_t)255; return p;
  };
  float*  scal    = (float*) take(64 * 4);
  bf16_t* xbf     = (bf16_t*)take(GN * IN_ * 2);
  bf16_t* h1bf    = (bf16_t*)take(GN * HID_ * 2);
  float*  zf      = (float*) take(GN * D_ * 4);
  bf16_t* encW1t  = (bf16_t*)take((size_t)HID_ * IN_ * 2);
  bf16_t* encW2t  = (bf16_t*)take((size_t)D_ * HID_ * 2);
  bf16_t* WQt     = (bf16_t*)take((size_t)D_ * D_ * 2);
  bf16_t* WKt     = (bf16_t*)take((size_t)D_ * D_ * 2);
  bf16_t* WQmt    = (bf16_t*)take((size_t)D_ * D_ * 2);
  bf16_t* WQb     = (bf16_t*)take((size_t)D_ * D_ * 2);
  bf16_t* WKb     = (bf16_t*)take((size_t)D_ * D_ * 2);
  bf16_t* WQmb    = (bf16_t*)take((size_t)D_ * D_ * 2);
  bf16_t* Kmb     = (bf16_t*)take((size_t)KM_ * D_ * 2);
  float*  deg     = (float*) take((size_t)G_ * T_ * 4);
  bf16_t* Mmat    = (bf16_t*)take((size_t)G_ * TP_ * TP_ * 2);
  float*  Vvec    = (float*) take(GT * 32 * 4);
  float*  xa      = (float*) take(GT * D_ * 4);
  bf16_t* gbf     = (bf16_t*)take(GT * D_ * 2);
  bf16_t* Qb      = (bf16_t*)take(GT * D_ * 2);
  bf16_t* Kb      = (bf16_t*)take(GT * D_ * 2);
  bf16_t* qmb     = (bf16_t*)take(GT * D_ * 2);
  bf16_t* Qtr     = (bf16_t*)take(GT * D_ * 2);
  bf16_t* Ktr     = (bf16_t*)take(GT * D_ * 2);
  bf16_t* dQbf    = (bf16_t*)take(GT * D_ * 2);
  bf16_t* dKbf    = (bf16_t*)take(GT * D_ * 2);
  bf16_t* dqmb    = (bf16_t*)take(GT * D_ * 2);
  float*  Smat    = (float*) take((size_t)G_ * H_ * TP_ * TP_ * 4);
  bf16_t* Pw      = (bf16_t*)take((size_t)G_ * H_ * TP_ * TP_ * 2);
  bf16_t* Wsym    = (bf16_t*)take((size_t)G_ * H_ * TP_ * TP_ * 2);
  float*  dgbuf   = (float*) take(GT * D_ * 4);
  if (off > ws_size) return;  // workspace too small: bail deterministically

  // padded rows of gbf must be zero (harness poisons ws)
  (void)hipMemsetAsync(gbf, 0, GT * D_ * 2, stream);

  // scalar prep + bf16 weight staging
  k_scalars<<<1, 32, 0, stream>>>(lam_a, lam_m, bet_a, bet_m, eta_l, scal);
  k_cvt  <<<1024, 256, 0, stream>>>(x, xbf, GN * IN_);
  k_cvt_t<<<64,   256, 0, stream>>>(enc_W1, encW1t, IN_, HID_);
  k_cvt_t<<<128,  256, 0, stream>>>(enc_W2, encW2t, HID_, D_);
  k_cvt_t<<<64,   256, 0, stream>>>(W_Q,  WQt,  D_, D_);
  k_cvt_t<<<64,   256, 0, stream>>>(W_K,  WKt,  D_, D_);
  k_cvt_t<<<64,   256, 0, stream>>>(W_Qm, WQmt, D_, D_);
  k_cvt  <<<64,   256, 0, stream>>>(W_Q,  WQb,  (size_t)D_ * D_);
  k_cvt  <<<64,   256, 0, stream>>>(W_K,  WKb,  (size_t)D_ * D_);
  k_cvt  <<<64,   256, 0, stream>>>(W_Qm, WQmb, (size_t)D_ * D_);
  k_km   <<<KM_, 128, 0, stream>>>(B_mem, W_Km, Kmb);

  // graph structure + Laplacian PE (block power iteration in 258 KB LDS)
  k_degree <<<2056, 256, 0, stream>>>(adj, deg);
  k_build_M<<<4096, 256, 0, stream>>>(adj, deg, Mmat);
  (void)hipFuncSetAttribute(reinterpret_cast<const void*>(k_power),
                            hipFuncAttributeMaxDynamicSharedMemorySize, (int)PWR_SMEM);
  k_power<<<G_, 256, PWR_SMEM, stream>>>(Mmat, Vvec);

  // node encoder: gelu(x W1 + b1) W2 + b2, then LN + PE inside k_assemble
  gemm_launch(stream, xbf, encW1t, nullptr, h1bf, enc_b1, nullptr,
              (int)GN, HID_, IN_, IN_, IN_, HID_, 1, 1, 0, 0, 0, 0, 0, 0, GF_GELU | GF_WBF);
  gemm_launch(stream, h1bf, encW2t, zf, nullptr, enc_b2, nullptr,
              (int)GN, D_, HID_, HID_, HID_, D_, 1, 1, 0, 0, 0, 0, 0, 0, GF_WF32);
  k_assemble<<<2056, 256, 0, stream>>>(zf, enc_lg, enc_lb, cls, Vvec, pe_W, pe_b, xa);

  // 8 energy-descent steps
  const long TT = (long)TP_ * TP_;
  for (int step = 0; step < NSTEP; ++step){
    k_ln<<<2056, 256, 0, stream>>>(xa, ln_g, ln_b, gbf);
    gemm_launch(stream, gbf, WQt,  nullptr, Qb,  nullptr, nullptr,
                (int)GT, D_, D_, D_, D_, D_, 1, 1, 0, 0, 0, 0, 0, 0, GF_WBF);
    gemm_launch(stream, gbf, WKt,  nullptr, Kb,  nullptr, nullptr,
                (int)GT, D_, D_, D_, D_, D_, 1, 1, 0, 0, 0, 0, 0, 0, GF_WBF);
    gemm_launch(stream, gbf, WQmt, nullptr, qmb, nullptr, nullptr,
                (int)GT, D_, D_, D_, D_, D_, 1, 1, 0, 0, 0, 0, 0, 0, GF_WBF);
    // per-head scores S = Qh Kh^T (K = HD = 32 -> one WMMA per accumulator)
    gemm_launch(stream, Qb, Kb, Smat, nullptr, nullptr, nullptr,
                TP_, TP_, HD_, D_, D_, TP_, G_, H_,
                (long)TP_ * D_, (long)TP_ * D_, (long)H_ * TT,
                HD_, HD_, TT, GF_WF32);
    k_softmax<<<8224, 256, 0, stream>>>(Smat, adj, scal, Pw);
    k_sym<<<4096, 256, 0, stream>>>(Pw, Wsym);
    k_qk_transpose<<<2048, 256, 0, stream>>>(Qb, Kb, Qtr, Ktr);
    // dQ = (lam/scale) (P+P^T) K ;  dK = (lam/scale) (P+P^T) Q
    gemm_launch(stream, Wsym, Ktr, nullptr, dQbf, nullptr, scal + 5,
                TP_, HD_, TP_, TP_, TP_, D_, G_, H_,
                (long)H_ * TT, (long)H_ * HD_ * TP_, (long)TP_ * D_,
                TT, (long)HD_ * TP_, HD_, GF_WBF);
    gemm_launch(stream, Wsym, Qtr, nullptr, dKbf, nullptr, scal + 5,
                TP_, HD_, TP_, TP_, TP_, D_, G_, H_,
                (long)H_ * TT, (long)H_ * HD_ * TP_, (long)TP_ * D_,
                TT, (long)HD_ * TP_, HD_, GF_WBF);
    k_mem<<<2304, 256, 0, stream>>>(qmb, Kmb, scal, dqmb);
    // dg = dQ W_Q^T + dK W_K^T + dqm W_Qm^T
    gemm_launch(stream, dQbf, WQb, dgbuf, nullptr, nullptr, nullptr,
                (int)GT, D_, D_, D_, D_, D_, 1, 1, 0, 0, 0, 0, 0, 0, GF_WF32);
    gemm_launch(stream, dKbf, WKb, dgbuf, nullptr, nullptr, nullptr,
                (int)GT, D_, D_, D_, D_, D_, 1, 1, 0, 0, 0, 0, 0, 0, GF_WF32 | GF_ACC);
    gemm_launch(stream, dqmb, WQmb, dgbuf, nullptr, nullptr, nullptr,
                (int)GT, D_, D_, D_, D_, D_, 1, 1, 0, 0, 0, 0, 0, 0, GF_WF32 | GF_ACC);
    k_update<<<2056, 256, 0, stream>>>(xa, dgbuf, ln_g, scal);
  }

  // graph readout on CLS tokens
  k_readout<<<G_, 128, 0, stream>>>(xa, ln_g, ln_b,
                                    rd_W1, rd_b1, rd_g1, rd_bb1,
                                    rd_W2, rd_b2, rd_g2, rd_bb2,
                                    rd_W3, rd_b3, out);
}
